// hypercorre_topk2_82008105550170
// MI455X (gfx1250) — matrix operand order repaired and checked by reading, not compiled, not run on hardware
//
#include <hip/hip_runtime.h>

// ---------------------------------------------------------------------------
// Fused window cross-attention for MI455X (gfx1250, wave32, WMMA).
//   q = (x Wq^T + bq) * d^-0.5 ; kv = mem Wkv^T + bkv
//   S = QK^T + rel-pos bias ; P = softmax(S) ; O = PV ; out = O Wp^T + bp
// All GEMMs run on v_wmma_f32_16x16x32_bf16 (bf16 in, f32 accumulate).
// One workgroup (16 waves) per batch window; one head per wave for attention.
// Projection GEMMs: each wave owns output-column tiles and accumulates all
// 4 M-tiles per weight B-fragment (register B reuse, 4x less L2 traffic).
// ---------------------------------------------------------------------------

typedef __attribute__((ext_vector_type(16))) __bf16 v16bf;
typedef __attribute__((ext_vector_type(8)))  float  v8f;

union BFVec { v16bf v; unsigned short u[16]; };

#define C_DIM   512
#define HEADS_N 16
#define HD      32
#define NTOK    49
#define TMEM    2
#define WIN_ELEMS (NTOK * C_DIM)   // 25088

__device__ __forceinline__ unsigned short f2bf(float f) {
  unsigned int x = __float_as_uint(f);
  x += 0x7FFFu + ((x >> 16) & 1u);          // round-to-nearest-even
  return (unsigned short)(x >> 16);
}

// A fragment: 16x32 bf16 tile, rows m = row0 + lane%16, row-major src (stride ld).
// Per lane: elements k0+h*16+{0..15} -> 16 contiguous bf16 (two b128 loads).
// Out-of-range rows read from a zeroed LDS pad (branchless, keeps EXEC intact).
__device__ __forceinline__ v16bf load_A_pad(const unsigned short* base, int row0, int k0,
                                            int ld, int rowlimit,
                                            const unsigned short* zpad, int lane) {
  const int m = row0 + (lane & 15);
  const unsigned short* r = base + m * ld + k0 + ((lane >> 4) << 4);
  if (m >= rowlimit) r = zpad;              // address select, no EXEC masking
  BFVec a;
#pragma unroll
  for (int j = 0; j < 8; ++j) a.u[j] = r[j];
#pragma unroll
  for (int j = 0; j < 8; ++j) a.u[8 + j] = r[8 + j];
  return a.v;
}

// A fragment, all rows known valid.
__device__ __forceinline__ v16bf load_A_nc(const unsigned short* base, int row0, int k0,
                                           int ld, int lane) {
  const unsigned short* r = base + (row0 + (lane & 15)) * ld + k0 + ((lane >> 4) << 4);
  BFVec a;
#pragma unroll
  for (int j = 0; j < 8; ++j) a.u[j] = r[j];
#pragma unroll
  for (int j = 0; j < 8; ++j) a.u[8 + j] = r[8 + j];
  return a.v;
}

// B fragment, transposed source: B[k][n] = W[o0+n][k0+k], W row-major stride ld.
// Lanes 0-15 hold K=0..15, lanes 16-31 hold K=16..31 (16 contiguous bf16 per lane).
__device__ __forceinline__ v16bf load_Bt(const unsigned short* W, int o0, int k0,
                                         int ld, int rowlimit, int lane) {
  const int n = lane & 15, hf = lane >> 4;
  int row = o0 + n;
  if (row >= rowlimit) row = 0;             // clamped rows are masked downstream
  const unsigned short* r = W + row * ld + k0 + (hf << 4);
  BFVec b;
#pragma unroll
  for (int j = 0; j < 16; ++j) b.u[j] = r[j];
  return b.v;
}

// B fragment, direct source: B[k][n] = V[k0+k][n0+n], V row-major stride ld.
__device__ __forceinline__ v16bf load_Bd(const unsigned short* V, int k0, int n0,
                                         int ld, int rowlimit, int lane) {
  const int n = lane & 15, hf = lane >> 4;
  BFVec b;
#pragma unroll
  for (int j = 0; j < 16; ++j) {
    const int k = k0 + (hf << 4) + j;
    b.u[j] = (k < rowlimit) ? V[k * ld + n0 + n] : (unsigned short)0;
  }
  return b.v;
}

// One output-column tile (16 cols at o0) x all 4 M-tiles; B loaded once per k-chunk.
__device__ __forceinline__ void mm_col(const unsigned short* act, const unsigned short* W,
                                       int o0, const unsigned short* zpad, int lane,
                                       v8f acc[4]) {
#pragma unroll 2
  for (int k0 = 0; k0 < C_DIM; k0 += 32) {
    v16bf b = load_Bt(W, o0, k0, C_DIM, 1 << 20, lane);
#pragma unroll
    for (int mt = 0; mt < 4; ++mt) {
      v16bf a = load_A_pad(act, mt * 16, k0, C_DIM, NTOK, zpad, lane);
      acc[mt] = __builtin_amdgcn_wmma_f32_16x16x32_bf16(false, a, false, b, (short)0,
                                                        acc[mt], false, false);
    }
  }
}

// ---------------------------------------------------------------------------
// Prep: f32 weights -> bf16 bits; gather rel-pos bias -> dense (H,49,49) f32.
// ---------------------------------------------------------------------------
__global__ void prep_kernel(const float* __restrict__ qw, const float* __restrict__ kvw,
                            const float* __restrict__ pw, const float* __restrict__ rpb,
                            const int* __restrict__ rel_idx,
                            unsigned short* __restrict__ qw_bf,
                            unsigned short* __restrict__ kvw_bf,
                            unsigned short* __restrict__ pw_bf,
                            float* __restrict__ bias_pre) {
  const int QW = 512 * 512, KVW = 1024 * 512, PW = 512 * 512;
  const int BIAS = HEADS_N * NTOK * NTOK;
  const int total = QW + KVW + PW + BIAS;
  for (int i = blockIdx.x * blockDim.x + threadIdx.x; i < total;
       i += gridDim.x * blockDim.x) {
    if (i < QW)                 qw_bf[i] = f2bf(qw[i]);
    else if (i < QW + KVW)      kvw_bf[i - QW] = f2bf(kvw[i - QW]);
    else if (i < QW + KVW + PW) pw_bf[i - QW - KVW] = f2bf(pw[i - QW - KVW]);
    else {
      const int j = i - (QW + KVW + PW);
      const int h = j / (NTOK * NTOK), mn = j % (NTOK * NTOK);
      bias_pre[j] = rpb[rel_idx[mn] * HEADS_N + h];
    }
  }
}

// ---------------------------------------------------------------------------
// Fused kernel: one block (16 waves) per batch window.
// ---------------------------------------------------------------------------
__global__ void __launch_bounds__(512, 1)
fused_window_attn(const float* __restrict__ x, const float* __restrict__ memory,
                  const float* __restrict__ q_b, const float* __restrict__ kv_b,
                  const float* __restrict__ proj_b,
                  const unsigned short* __restrict__ qw_bf,
                  const unsigned short* __restrict__ kvw_bf,
                  const unsigned short* __restrict__ pw_bf,
                  const float* __restrict__ bias_pre,
                  float* __restrict__ out) {
  extern __shared__ unsigned short smem[];
  unsigned short* sAct = smem;                      // 49x512 bf16 (x, then mem[t])
  unsigned short* sQ   = sAct + WIN_ELEMS;          // 49x512 scaled q
  unsigned short* sK   = sQ   + WIN_ELEMS;          // 49x512 k
  unsigned short* sV   = sK   + WIN_ELEMS;          // 49x512 v
  unsigned short* sO   = sV   + WIN_ELEMS;          // 49x512 attention out
  unsigned short* sP   = sO   + WIN_ELEMS;          // 16 waves x 16x64 prob scratch
  unsigned short* zpad = sP + 16 * 1024;            // 32 zeros for padded A rows

  const int tid  = threadIdx.x;
  const int wave = tid >> 5;
  const int lane = tid & 31;
  const int bb   = blockIdx.x;
  const float scale = 0.17677669529663687f;         // 1/sqrt(32)

  if (tid < 32) zpad[tid] = 0;

  // ---- stage x[bb] -> bf16 LDS ----
  {
    const float4* src = (const float4*)(x + (size_t)bb * WIN_ELEMS);
    for (int i = tid; i < WIN_ELEMS / 4; i += 512) {
      float4 f = src[i];
      const int o = i * 4;
      sAct[o+0]=f2bf(f.x); sAct[o+1]=f2bf(f.y); sAct[o+2]=f2bf(f.z); sAct[o+3]=f2bf(f.w);
    }
  }
  __syncthreads();

  // ---- q projection: 32 col-tiles over 16 waves, 2 each ----
#pragma unroll
  for (int i = 0; i < 2; ++i) {
    const int ot = wave + 16 * i;
    v8f acc[4] = {};
    mm_col(sAct, qw_bf, ot * 16, zpad, lane, acc);
    const int n = lane & 15, hf = lane >> 4;
    const float bq = q_b[ot * 16 + n];
#pragma unroll
    for (int mt = 0; mt < 4; ++mt)
#pragma unroll
      for (int r = 0; r < 8; ++r) {
        const int m = mt * 16 + r + hf * 8;
        if (m < NTOK) sQ[m * C_DIM + ot * 16 + n] = f2bf((acc[mt][r] + bq) * scale);
      }
  }
  __syncthreads();

  for (int t = 0; t < TMEM; ++t) {
    // ---- stage memory[bb*T + t] ----
    {
      const float4* src = (const float4*)(memory + (size_t)(bb * TMEM + t) * WIN_ELEMS);
      for (int i = tid; i < WIN_ELEMS / 4; i += 512) {
        float4 f = src[i];
        const int o = i * 4;
        sAct[o+0]=f2bf(f.x); sAct[o+1]=f2bf(f.y); sAct[o+2]=f2bf(f.z); sAct[o+3]=f2bf(f.w);
      }
    }
    __syncthreads();

    // ---- kv projection: 64 col-tiles over 16 waves; o<512 -> K, o>=512 -> V ----
#pragma unroll
    for (int i = 0; i < 4; ++i) {
      const int ot = wave + 16 * i;
      v8f acc[4] = {};
      mm_col(sAct, kvw_bf, ot * 16, zpad, lane, acc);
      const int n = lane & 15, hf = lane >> 4;
      const int o = ot * 16 + n;
      const float bkv = kv_b[o];
      unsigned short* dst = (o < C_DIM) ? sK : sV;
      const int oc = o & (C_DIM - 1);
#pragma unroll
      for (int mt = 0; mt < 4; ++mt)
#pragma unroll
        for (int r = 0; r < 8; ++r) {
          const int m = mt * 16 + r + hf * 8;
          if (m < NTOK) dst[m * C_DIM + oc] = f2bf(acc[mt][r] + bkv);
        }
    }
    __syncthreads();

    // ---- attention: one head per wave ----
    {
      const int h = wave;
      const int hcol = h * HD;
      const float* biasH = bias_pre + h * NTOK * NTOK;
      unsigned short* myP = sP + wave * 16 * 64;
      const int n = lane & 15, hf = lane >> 4;

      v16bf Bk[4];                                   // K^T fragments (tokens as N)
#pragma unroll
      for (int nt = 0; nt < 4; ++nt)
        Bk[nt] = load_Bt(sK, nt * 16, hcol, C_DIM, NTOK, lane);
      v16bf Bv[2][2];                                // V fragments (tokens as K)
#pragma unroll
      for (int kk = 0; kk < 2; ++kk)
#pragma unroll
        for (int dt = 0; dt < 2; ++dt)
          Bv[kk][dt] = load_Bd(sV, kk * 32, hcol + dt * 16, C_DIM, NTOK, lane);

      for (int mt = 0; mt < 4; ++mt) {
        v16bf Aq = load_A_pad(sQ, mt * 16, hcol, C_DIM, NTOK, zpad, lane);
        v8f S[4];
#pragma unroll
        for (int nt = 0; nt < 4; ++nt) {
          v8f z = {};
          S[nt] = __builtin_amdgcn_wmma_f32_16x16x32_bf16(false, Aq, false, Bk[nt],
                                                          (short)0, z, false, false);
        }
        // bias + key-padding mask
#pragma unroll
        for (int nt = 0; nt < 4; ++nt) {
          const int col = nt * 16 + n;
#pragma unroll
          for (int r = 0; r < 8; ++r) {
            const int m = mt * 16 + r + hf * 8;
            float s = S[nt][r];
            if (col < NTOK) s += (m < NTOK) ? biasH[m * NTOK + col] : 0.0f;
            else            s = -1e30f;
            S[nt][r] = s;
          }
        }
        // softmax over keys: each (r, lane-half) is one row across 16 lanes x 4 tiles
#pragma unroll
        for (int r = 0; r < 8; ++r) {
          float mx = fmaxf(fmaxf(S[0][r], S[1][r]), fmaxf(S[2][r], S[3][r]));
          for (int off = 8; off; off >>= 1) mx = fmaxf(mx, __shfl_xor(mx, off, 16));
          float sum = 0.0f;
#pragma unroll
          for (int nt = 0; nt < 4; ++nt) {
            const float e = __expf(S[nt][r] - mx);
            S[nt][r] = e; sum += e;
          }
          for (int off = 8; off; off >>= 1) sum += __shfl_xor(sum, off, 16);
          const float inv = 1.0f / sum;
#pragma unroll
          for (int nt = 0; nt < 4; ++nt) S[nt][r] *= inv;
        }
        // transpose P through wave-private LDS scratch (D layout -> A layout)
#pragma unroll
        for (int nt = 0; nt < 4; ++nt)
#pragma unroll
          for (int r = 0; r < 8; ++r)
            myP[(r + hf * 8) * 64 + nt * 16 + n] = f2bf(S[nt][r]);
        asm volatile("s_wait_dscnt 0x0" ::: "memory");
        // O tile: P(16x64) x V(64x32)
#pragma unroll
        for (int dt = 0; dt < 2; ++dt) {
          v8f Ov = {};
#pragma unroll
          for (int kk = 0; kk < 2; ++kk) {
            v16bf Ap = load_A_nc(myP, 0, kk * 32, 64, lane);
            Ov = __builtin_amdgcn_wmma_f32_16x16x32_bf16(false, Ap, false, Bv[kk][dt],
                                                         (short)0, Ov, false, false);
          }
#pragma unroll
          for (int r = 0; r < 8; ++r) {
            const int m = mt * 16 + r + hf * 8;
            if (m < NTOK) sO[m * C_DIM + hcol + dt * 16 + n] = f2bf(Ov[r]);
          }
        }
      }
    }
    __syncthreads();

    // ---- output projection -> global (f32) ----
    float* outp = out + (size_t)(bb * TMEM + t) * WIN_ELEMS;
#pragma unroll
    for (int i = 0; i < 2; ++i) {
      const int ot = wave + 16 * i;
      v8f acc[4] = {};
      mm_col(sO, pw_bf, ot * 16, zpad, lane, acc);
      const int n = lane & 15, hf = lane >> 4;
      const int o = ot * 16 + n;
      const float bp = proj_b[o];
#pragma unroll
      for (int mt = 0; mt < 4; ++mt)
#pragma unroll
        for (int r = 0; r < 8; ++r) {
          const int m = mt * 16 + r + hf * 8;
          if (m < NTOK) outp[m * C_DIM + o] = acc[mt][r] + bp;
        }
    }
    __syncthreads();
  }
}

// ---------------------------------------------------------------------------
extern "C" void kernel_launch(void* const* d_in, const int* in_sizes, int n_in,
                              void* d_out, int out_size, void* d_ws, size_t ws_size,
                              hipStream_t stream) {
  (void)in_sizes; (void)n_in; (void)out_size; (void)ws_size;
  const float* x      = (const float*)d_in[0];
  const float* memory = (const float*)d_in[1];
  const float* rpb    = (const float*)d_in[2];
  const float* q_w    = (const float*)d_in[3];
  const float* q_b    = (const float*)d_in[4];
  const float* kv_w   = (const float*)d_in[5];
  const float* kv_b   = (const float*)d_in[6];
  const float* proj_w = (const float*)d_in[7];
  const float* proj_b = (const float*)d_in[8];
  const int*   rel_idx= (const int*)d_in[9];
  float* out = (float*)d_out;

  // workspace: bf16 weights (2 MB) + dense bias table (150 KB)
  unsigned short* qw_bf  = (unsigned short*)d_ws;
  unsigned short* kvw_bf = qw_bf + 512 * 512;
  unsigned short* pw_bf  = kvw_bf + 1024 * 512;
  float* bias_pre = (float*)(pw_bf + 512 * 512);

  prep_kernel<<<1024, 256, 0, stream>>>(q_w, kv_w, proj_w, rpb, rel_idx,
                                        qw_bf, kvw_bf, pw_bf, bias_pre);

  const int SMEM = (5 * WIN_ELEMS + 16 * 1024 + 32) * (int)sizeof(unsigned short);
  (void)hipFuncSetAttribute(reinterpret_cast<const void*>(fused_window_attn),
                            hipFuncAttributeMaxDynamicSharedMemorySize, SMEM);
  fused_window_attn<<<2048, 512, SMEM, stream>>>(x, memory, q_b, kv_b, proj_b,
                                                 qw_bf, kvw_bf, pw_bf, bias_pre, out);
}
